// MHA_27049704030797
// MI455X (gfx1250) — compile-verified, matrix operand order checked by reading, not run on hardware
//
#include <hip/hip_runtime.h>
#include <hip/hip_bf16.h>

// ---------------------------------------------------------------------------
// MHA forward, fused flash-attention, all matmuls via v_wmma_f32_16x16x32_f16;
// attention K/V tiles streamed HBM->LDS with GLOBAL_LOAD_ASYNC_TO_LDS_B128
// (double-buffered, ASYNCcnt-pipelined).
// B=2, N=M=2048, C=D=1024, H=16, hd=64
// ---------------------------------------------------------------------------

typedef __attribute__((ext_vector_type(16))) _Float16 v16h;
typedef __attribute__((ext_vector_type(8)))  float    v8f;

union F16Frag { v16h v; unsigned int u[8]; };

// WMMA f32 = f16(16x32) x f16(32x16) + f32(16x16)
__device__ __forceinline__ v8f wmma32(v16h a, v16h b, v8f c) {
  return __builtin_amdgcn_wmma_f32_16x16x32_f16(
      false, a, false, b, (short)0, c, false, false);
}

// Load a 16x32 f16 A-fragment (or the mirrored B-fragment from an N-major
// tile) from memory laid out row-major with `stride` halfs per row.
// ISA 7.12.2: lanes 0-15 hold row=lane, K={0..7,16..23}; lanes 16-31 hold
// row=lane-16, K={8..15,24..31}; 2 halfs per VGPR.
__device__ __forceinline__ v16h load_frag16(const _Float16* p, int stride) {
  const int lane = threadIdx.x & 31;
  const int row  = lane & 15;
  const int kb   = (lane & 16) ? 8 : 0;
  const _Float16* rp = p + row * stride;
  F16Frag f;
#pragma unroll
  for (int v = 0; v < 8; ++v) {
    const int k = (v < 4) ? (kb + 2 * v) : (16 + kb + 2 * (v - 4));
    f.u[v] = *reinterpret_cast<const unsigned int*>(rp + k);
  }
  return f.v;
}

// Generic __shared__ pointer -> LDS byte offset (ISA 10.2: LDS aperture maps
// addr[31:0] to the LDS offset).
__device__ __forceinline__ unsigned lds_off(const void* p) {
  return (unsigned)(unsigned long long)p;
}

// Async HBM -> LDS 16B copy (per lane), tracked by ASYNCcnt.
__device__ __forceinline__ void async_ldg_b128(unsigned lds_addr,
                                               const _Float16* g) {
  asm volatile("global_load_async_to_lds_b128 %0, %1, off"
               :: "v"(lds_addr), "v"(g) : "memory");
}

// ---------------------------------------------------------------------------
// Kernel 1: QKV projections.  act[B*S,1024] (f32) @ W[1024,1024] (f32) + bias
// -> f16, written as:
//   proj 0 (K): Kb[b][h][m][d]      row-major   (B-frag source for Q.K^T)
//   proj 1 (Q): Qb[b][h][n][d]      row-major   (A-frag source)
//   proj 2 (V): Vt[b][h][d][m]      TRANSPOSED  (B-frag source for P.V)
// Workgroup tile: 128 rows x 64 cols, 8 waves of 32x32.  (f32->f16 conversion
// happens during the LDS fill, so this path stays on VALU loads.)
// ---------------------------------------------------------------------------
__global__ __launch_bounds__(256) void qkv_proj_kernel(
    const float* __restrict__ kv, const float* __restrict__ qin,
    const float* __restrict__ Wk, const float* __restrict__ bk,
    const float* __restrict__ Wq, const float* __restrict__ bq,
    const float* __restrict__ Wv, const float* __restrict__ bv,
    _Float16* __restrict__ Kb, _Float16* __restrict__ Qb,
    _Float16* __restrict__ Vt)
{
  const int proj = blockIdx.z;
  const float* A    = (proj == 1) ? qin : kv;
  const float* W    = (proj == 0) ? Wk : ((proj == 1) ? Wq : Wv);
  const float* bias = (proj == 0) ? bk : ((proj == 1) ? bq : bv);

  __shared__ _Float16 Alds[128 * 32];  // act tile, row-major
  __shared__ _Float16 Wlds[64 * 32];   // weight tile, N-major (transposed)

  const int tid  = threadIdx.x;
  const int lane = tid & 31;
  const int wave = tid >> 5;
  const int wr   = wave & 3;   // 4 row groups of 32
  const int wc   = wave >> 2;  // 2 col groups of 32
  const int rowbase = blockIdx.x * 128;
  const int colbase = blockIdx.y * 64;

  v8f acc[2][2] = {};

  for (int k0 = 0; k0 < 1024; k0 += 32) {
    __syncthreads();
#pragma unroll
    for (int i = 0; i < 16; ++i) {
      const int idx = tid + i * 256;          // 128x32 elements
      const int r = idx >> 5, k = idx & 31;
      Alds[idx] = (_Float16)A[(size_t)(rowbase + r) * 1024 + k0 + k];
    }
#pragma unroll
    for (int i = 0; i < 8; ++i) {
      const int idx = tid + i * 256;          // 32x64 elements, transpose
      const int k = idx >> 6, n = idx & 63;
      Wlds[n * 32 + k] = (_Float16)W[(size_t)(k0 + k) * 1024 + colbase + n];
    }
    __syncthreads();

    v16h af[2], bf[2];
    af[0] = load_frag16(&Alds[(wr * 32 +  0) * 32], 32);
    af[1] = load_frag16(&Alds[(wr * 32 + 16) * 32], 32);
    bf[0] = load_frag16(&Wlds[(wc * 32 +  0) * 32], 32);
    bf[1] = load_frag16(&Wlds[(wc * 32 + 16) * 32], 32);
#pragma unroll
    for (int i = 0; i < 2; ++i)
#pragma unroll
      for (int j = 0; j < 2; ++j)
        acc[i][j] = wmma32(af[i], bf[j], acc[i][j]);
  }

  const int colloc = lane & 15;
  const int rowoff = (lane & 16) ? 8 : 0;
#pragma unroll
  for (int i = 0; i < 2; ++i) {
#pragma unroll
    for (int j = 0; j < 2; ++j) {
      const int col = colbase + wc * 32 + j * 16 + colloc;
      const float bconst = bias[col];
      const int h = col >> 6, d = col & 63;
#pragma unroll
      for (int r = 0; r < 8; ++r) {
        const int row = rowbase + wr * 32 + i * 16 + r + rowoff;
        const int b = row >> 11, s = row & 2047;
        const float val = acc[i][j][r] + bconst;
        if (proj == 2) {
          Vt[(((size_t)(b * 16 + h)) * 64 + d) * 2048 + s] = (_Float16)val;
        } else {
          _Float16* dst = (proj == 0) ? Kb : Qb;
          dst[(((size_t)(b * 16 + h)) * 2048 + s) * 64 + d] = (_Float16)val;
        }
      }
    }
  }
}

// ---------------------------------------------------------------------------
// Kernel 2: flash attention.  grid = (N/128, B*H).  8 waves, each owns 16
// query rows; Q fragments stay in registers; K/V streamed in 32-wide tiles
// through double-buffered LDS via GLOBAL_LOAD_ASYNC_TO_LDS_B128 (copy of
// tile mt+1 overlaps the 8 WMMAs + online softmax of tile mt); P restaged
// through per-wave LDS for the P.V WMMAs.  hd = 64, scale = 1/8.
// ---------------------------------------------------------------------------
__global__ __launch_bounds__(256) void attn_kernel(
    const _Float16* __restrict__ Qb, const _Float16* __restrict__ Kb,
    const _Float16* __restrict__ Vt, _Float16* __restrict__ Yb)
{
  __shared__ __align__(16) _Float16 Klds[2][32 * 64];  // K rows (m) x hd
  __shared__ __align__(16) _Float16 Vlds[2][64 * 32];  // Vt rows (d) x m
  __shared__ __align__(16) _Float16 Plds[8 * 16 * 32]; // per-wave P staging

  const int tid  = threadIdx.x;
  const int lane = tid & 31;
  const int wave = tid >> 5;
  const int bh   = blockIdx.y;
  const int q0   = blockIdx.x * 128 + wave * 16;

  const _Float16* Qrow = Qb + ((size_t)bh * 2048 + q0) * 64;
  const v16h aq0 = load_frag16(Qrow, 64);       // d = 0..31
  const v16h aq1 = load_frag16(Qrow + 32, 64);  // d = 32..63

  float mrun[8], lrun[8];
  v8f acc[4] = {};
#pragma unroll
  for (int r = 0; r < 8; ++r) { mrun[r] = -1e30f; lrun[r] = 0.0f; }

  const int colloc = lane & 15;
  const int rowoff = (lane & 16) ? 8 : 0;
  _Float16* Pl = &Plds[wave * 512];

  // Per-thread async copy sources: each thread moves one 16B chunk of the
  // K tile (contiguous 32x64 halfs) and one 16B chunk of the V tile
  // (64 rows (d) x 32 halfs, 4 chunks per row).
  const _Float16* Kg = Kb + (size_t)bh * 2048 * 64;  // + m0*64 + tid*8
  const int vd = tid >> 2, vseg = tid & 3;
  const _Float16* Vg = Vt + ((size_t)bh * 64 + vd) * 2048 + vseg * 8;  // + m0
  const unsigned kdst0 = lds_off(&Klds[0][tid * 8]);
  const unsigned kdst1 = lds_off(&Klds[1][tid * 8]);
  const unsigned vdst0 = lds_off(&Vlds[0][vd * 32 + vseg * 8]);
  const unsigned vdst1 = lds_off(&Vlds[1][vd * 32 + vseg * 8]);

  // Preload tile 0 into buffer 0.
  async_ldg_b128(kdst0, Kg + (size_t)0 * 64 + tid * 8);
  async_ldg_b128(vdst0, Vg + 0);

  for (int mt = 0; mt < 64; ++mt) {
    const int m0  = mt * 32;
    const int buf = mt & 1;

    if (mt + 1 < 64) {  // overlap: issue tile mt+1 into the other buffer
      const int m1 = m0 + 32;
      async_ldg_b128(buf ? kdst0 : kdst1, Kg + (size_t)m1 * 64 + tid * 8);
      async_ldg_b128(buf ? vdst0 : vdst1, Vg + m1);
      // 2 ops of tile mt+1 stay in flight; wait only for tile mt's 2 ops.
      asm volatile("s_wait_asynccnt 0x2" ::: "memory");
    } else {
      asm volatile("s_wait_asynccnt 0x0" ::: "memory");
    }
    __syncthreads();  // all waves' tile-mt chunks visible in LDS

    const _Float16* Kt = Klds[buf];
    const _Float16* Vl = Vlds[buf];

    // S = Q.K^T : two 16x16 column tiles, contraction over hd=64 (2 k-steps)
    const v16h k00 = load_frag16(&Kt[0], 64);
    const v16h k01 = load_frag16(&Kt[32], 64);
    const v16h k10 = load_frag16(&Kt[16 * 64], 64);
    const v16h k11 = load_frag16(&Kt[16 * 64 + 32], 64);
    v8f s0 = {}, s1 = {};
    s0 = wmma32(aq0, k00, s0);  s0 = wmma32(aq1, k01, s0);
    s1 = wmma32(aq0, k10, s1);  s1 = wmma32(aq1, k11, s1);

    // Online softmax per row; C-layout => row stats reduce over 16-lane group
    float p0[8], p1[8];
#pragma unroll
    for (int r = 0; r < 8; ++r) {
      const float x0 = s0[r] * 0.125f;
      const float x1 = s1[r] * 0.125f;
      float mx = fmaxf(x0, x1);
      mx = fmaxf(mx, __shfl_xor(mx, 1, 32));
      mx = fmaxf(mx, __shfl_xor(mx, 2, 32));
      mx = fmaxf(mx, __shfl_xor(mx, 4, 32));
      mx = fmaxf(mx, __shfl_xor(mx, 8, 32));
      const float mnew = fmaxf(mrun[r], mx);
      const float corr = __expf(mrun[r] - mnew);
      mrun[r] = mnew;
      const float e0 = __expf(x0 - mnew);
      const float e1 = __expf(x1 - mnew);
      p0[r] = e0; p1[r] = e1;
      float rs = e0 + e1;
      rs += __shfl_xor(rs, 1, 32);
      rs += __shfl_xor(rs, 2, 32);
      rs += __shfl_xor(rs, 4, 32);
      rs += __shfl_xor(rs, 8, 32);
      lrun[r] = lrun[r] * corr + rs;
#pragma unroll
      for (int t = 0; t < 4; ++t) acc[t][r] *= corr;
    }

    // Restage P (C-layout -> row-major 16x32) in this wave's LDS region
#pragma unroll
    for (int r = 0; r < 8; ++r) {
      Pl[(r + rowoff) * 32 + colloc]      = (_Float16)p0[r];
      Pl[(r + rowoff) * 32 + 16 + colloc] = (_Float16)p1[r];
    }
    asm volatile("s_wait_dscnt 0" ::: "memory");  // same-wave LDS RAW

    const v16h pf = load_frag16(Pl, 32);
#pragma unroll
    for (int t = 0; t < 4; ++t) {
      const v16h vf = load_frag16(&Vl[t * 16 * 32], 32);
      acc[t] = wmma32(pf, vf, acc[t]);
    }
    __syncthreads();  // all waves done reading buf before it is re-filled
  }

  const int b = bh >> 4, h = bh & 15;
#pragma unroll
  for (int t = 0; t < 4; ++t) {
#pragma unroll
    for (int r = 0; r < 8; ++r) {
      const int row = q0 + r + rowoff;
      const int d   = h * 64 + t * 16 + colloc;
      Yb[((size_t)b * 2048 + row) * 1024 + d] =
          (_Float16)(acc[t][r] / lrun[r]);
    }
  }
}

// ---------------------------------------------------------------------------
// Kernel 3: output projection.  Y[B*N,1024] (f16) @ Wp (f32) + bp -> f32 out
// ---------------------------------------------------------------------------
__global__ __launch_bounds__(256) void out_proj_kernel(
    const _Float16* __restrict__ Y, const float* __restrict__ Wp,
    const float* __restrict__ bp, float* __restrict__ out)
{
  __shared__ _Float16 Alds[128 * 32];
  __shared__ _Float16 Wlds[64 * 32];

  const int tid  = threadIdx.x;
  const int lane = tid & 31;
  const int wave = tid >> 5;
  const int wr   = wave & 3;
  const int wc   = wave >> 2;
  const int rowbase = blockIdx.x * 128;
  const int colbase = blockIdx.y * 64;

  v8f acc[2][2] = {};

  for (int k0 = 0; k0 < 1024; k0 += 32) {
    __syncthreads();
#pragma unroll
    for (int i = 0; i < 8; ++i) {   // 128x32 f16 = 2048 dwords
      const int idx = tid + i * 256;
      const int r = idx >> 4, ku = idx & 15;
      ((unsigned int*)Alds)[idx] =
          ((const unsigned int*)(Y + (size_t)(rowbase + r) * 1024 + k0))[ku];
    }
#pragma unroll
    for (int i = 0; i < 8; ++i) {   // 32x64 f32 -> f16, transpose
      const int idx = tid + i * 256;
      const int k = idx >> 6, n = idx & 63;
      Wlds[n * 32 + k] = (_Float16)Wp[(size_t)(k0 + k) * 1024 + colbase + n];
    }
    __syncthreads();

    v16h af[2], bf[2];
    af[0] = load_frag16(&Alds[(wr * 32 +  0) * 32], 32);
    af[1] = load_frag16(&Alds[(wr * 32 + 16) * 32], 32);
    bf[0] = load_frag16(&Wlds[(wc * 32 +  0) * 32], 32);
    bf[1] = load_frag16(&Wlds[(wc * 32 + 16) * 32], 32);
#pragma unroll
    for (int i = 0; i < 2; ++i)
#pragma unroll
      for (int j = 0; j < 2; ++j)
        acc[i][j] = wmma32(af[i], bf[j], acc[i][j]);
  }

  const int colloc = lane & 15;
  const int rowoff = (lane & 16) ? 8 : 0;
#pragma unroll
  for (int i = 0; i < 2; ++i)
#pragma unroll
    for (int j = 0; j < 2; ++j) {
      const int col = colbase + wc * 32 + j * 16 + colloc;
      const float bconst = bp[col];
#pragma unroll
      for (int r = 0; r < 8; ++r) {
        const int row = rowbase + wr * 32 + i * 16 + r + rowoff;
        out[(size_t)row * 1024 + col] = acc[i][j][r] + bconst;
      }
    }
}

// ---------------------------------------------------------------------------
extern "C" void kernel_launch(void* const* d_in, const int* in_sizes, int n_in,
                              void* d_out, int out_size, void* d_ws,
                              size_t ws_size, hipStream_t stream) {
  (void)in_sizes; (void)n_in; (void)out_size; (void)ws_size;
  const float* kv = (const float*)d_in[0];
  const float* q  = (const float*)d_in[1];
  const float* Wk = (const float*)d_in[2];
  const float* bk = (const float*)d_in[3];
  const float* Wq = (const float*)d_in[4];
  const float* bq = (const float*)d_in[5];
  const float* Wv = (const float*)d_in[6];
  const float* bv = (const float*)d_in[7];
  const float* Wp = (const float*)d_in[8];
  const float* bp = (const float*)d_in[9];

  _Float16* ws = (_Float16*)d_ws;
  const size_t SZ = (size_t)2 * 16 * 2048 * 64;  // 4M halfs per tensor
  _Float16* Kb = ws;
  _Float16* Qb = ws + SZ;
  _Float16* Vt = ws + 2 * SZ;
  _Float16* Yb = ws + 3 * SZ;

  // QKV projections: 3 x (4096x1024x1024) GEMMs, f16 WMMA
  qkv_proj_kernel<<<dim3(32, 16, 3), 256, 0, stream>>>(
      kv, q, Wk, bk, Wq, bq, Wv, bv, Kb, Qb, Vt);
  // Fused flash attention: grid (N/128, B*H), async double-buffered K/V
  attn_kernel<<<dim3(16, 32), 256, 0, stream>>>(Qb, Kb, Vt, Yb);
  // Output projection -> f32
  out_proj_kernel<<<dim3(32, 16), 256, 0, stream>>>(Yb, Wp, bp,
                                                    (float*)d_out);
}
